// LPKT_28544352649372
// MI455X (gfx1250) — matrix operand order repaired and checked by reading, not compiled, not run on hardware
//
#include <hip/hip_runtime.h>

#define DK 128   // hidden dim
#define BB 128   // batch
#define SS 128   // sequence length
#define NQ 101   // N_Q + 1
#define TT 127   // scan steps (S-1)

typedef __attribute__((ext_vector_type(16))) __bf16 v16bf;
typedef __attribute__((ext_vector_type(8)))  __bf16 v8bf;
typedef __attribute__((ext_vector_type(8)))  float  v8f;

static __device__ __forceinline__ v8f wmma_bf(v16bf a, v16bf b, v8f c) {
    // (neg_a, A, neg_b, B, c_mod, C, reuse_a, reuse_b)
    return __builtin_amdgcn_wmma_f32_16x16x32_bf16(false, a, false, b, (short)0, c, false, false);
}
static __device__ __forceinline__ float sigf(float x) { return 1.0f / (1.0f + __expf(-x)); }

static __device__ __forceinline__ v16bf packA(v8bf c0, v8bf c1) {
    v16bf a;
#pragma unroll
    for (int i = 0; i < 8; i++) { a[i] = c0[i]; a[i + 8] = c1[i]; }
    return a;
}
static __device__ __forceinline__ v8bf cvt8(const float* p) {
    v8f f = *(const v8f*)p;
    v8bf r;
#pragma unroll
    for (int i = 0; i < 8; i++) r[i] = (__bf16)f[i];
    return r;
}
// A-operand (16x32 bf16): lanes 0-15 -> M=lane, k chunks {0..7,16..23}; lanes 16-31 -> +8
static __device__ __forceinline__ v16bf afrag_bf(const __bf16* A, int ld, int ks, int lane) {
    const __bf16* row = A + (lane & 15) * ld + ks * 32 + ((lane & 16) ? 8 : 0);
    return packA(*(const v8bf*)row, *(const v8bf*)(row + 16));
}
// B-operand (32x16 bf16) from row-major weight W[n][k] (B[k][n]=W[n][k]):
// lane%16 = n, k chunks {0..7,16..23}; lanes 16-31 -> +8
static __device__ __forceinline__ v16bf bfrag(const __bf16* W, int ld, int coloff, int nt, int ks, int lane) {
    int n = nt * 16 + (lane & 15);
    int k = coloff + ks * 32 + ((lane & 16) ? 8 : 0);
    return packA(*(const v8bf*)(W + n * ld + k), *(const v8bf*)(W + n * ld + k + 16));
}

// ---------------- small utility kernels ----------------
__global__ void k_cvt(const float* __restrict__ s, __bf16* __restrict__ d, int n) {
    int i = blockIdx.x * 256 + threadIdx.x;
    if (i < n) d[i] = (__bf16)s[i];
}
__global__ void k_c1(const float* __restrict__ W1, float* __restrict__ c1) {
    int o = threadIdx.x;
    float s = 0.f;
    for (int k = 0; k < 128; k++) s += W1[o * 384 + 256 + k];
    c1[o] = s;
}
__global__ void k_hinit(const float* __restrict__ h0, float* __restrict__ h) {
    int i = blockIdx.x * 256 + threadIdx.x;
    if (i < BB * NQ * DK) h[i] = h0[i % (NQ * DK)];
}

// ---------------- K1: all_learning = [e_emb|at_emb]@W1[:,:256].T + a*c1 + b1 (bf16 out) ----------------
__global__ __launch_bounds__(256) void k1_AL(
    const int* __restrict__ e_data, const int* __restrict__ at_data, const int* __restrict__ a_data,
    const float* __restrict__ e_w, const float* __restrict__ at_w,
    const __bf16* __restrict__ W1b, const float* __restrict__ b1, const float* __restrict__ c1,
    __bf16* __restrict__ AL)
{
    int lane = threadIdx.x & 31, wid = threadIdx.x >> 5;
    int half8 = (lane & 16) ? 8 : 0;
    int m = lane & 15;
    int rt = blockIdx.x;
    int r = rt * 16 + m;
    const float* erow = e_w + e_data[r] * DK;
    const float* arow = at_w + at_data[r] * DK;
    v16bf af[8];
#pragma unroll
    for (int ks = 0; ks < 8; ks++) {
        int klo = ks * 32 + half8;
        int khi = klo + 16;
        const float* p0 = (klo < 128) ? (erow + klo) : (arow + klo - 128);
        const float* p1 = (khi < 128) ? (erow + khi) : (arow + khi - 128);
        af[ks] = packA(cvt8(p0), cvt8(p1));
    }
    int nt = wid;
    v8f acc = {};
#pragma unroll
    for (int ks = 0; ks < 8; ks++) acc = wmma_bf(af[ks], bfrag(W1b, 384, 0, nt, ks, lane), acc);
    int c = nt * 16 + m;
    float bc = b1[c], cc = c1[c];
#pragma unroll
    for (int j = 0; j < 8; j++) {
        int row = rt * 16 + j + half8;
        AL[row * DK + c] = (__bf16)(acc[j] + bc + (float)a_data[row] * cc);
    }
}

// ---------------- K2: time-independent precomputes P2,P3,P4,E5 ----------------
__global__ __launch_bounds__(256) void k2_pre(
    const int* __restrict__ e_data, const int* __restrict__ it_data,
    const float* __restrict__ e_w, const float* __restrict__ it_w,
    const __bf16* __restrict__ AL,
    const __bf16* __restrict__ W2b, const __bf16* __restrict__ W3b,
    const __bf16* __restrict__ W4b, const __bf16* __restrict__ W5b,
    const float* __restrict__ b2, const float* __restrict__ b3,
    const float* __restrict__ b4, const float* __restrict__ b5,
    float* __restrict__ P2, float* __restrict__ P3,
    float* __restrict__ P4, float* __restrict__ E5)
{
    int lane = threadIdx.x & 31, wid = threadIdx.x >> 5;
    int half8 = (lane & 16) ? 8 : 0;
    int m = lane & 15;
    int rt = blockIdx.x;
    int r = rt * 16 + m;
    int b = r / TT, t = r % TT;
    const __bf16* alt  = AL + (b * SS + t) * DK;
    const __bf16* alm1 = (t > 0) ? (AL + (b * SS + t - 1) * DK) : (const __bf16*)0;
    const float* itrow = it_w + it_data[b * SS + t] * DK;
    const float* enrow = e_w + e_data[b * SS + t + 1] * DK;
    v8bf zero8;
#pragma unroll
    for (int i = 0; i < 8; i++) zero8[i] = (__bf16)0.0f;

    // z-prefix A frags (K=384): [AL[t-1] | it_emb | AL[t]]
    v16bf zf[12];
#pragma unroll
    for (int ks = 0; ks < 12; ks++) {
        int klo = ks * 32 + half8, khi = klo + 16;
        v8bf c0, c1v;
        if (klo < 128)      { c0 = alm1 ? *(const v8bf*)(alm1 + klo) : zero8; }
        else if (klo < 256) { c0 = cvt8(itrow + klo - 128); }
        else                { c0 = *(const v8bf*)(alt + klo - 256); }
        if (khi < 128)      { c1v = alm1 ? *(const v8bf*)(alm1 + khi) : zero8; }
        else if (khi < 256) { c1v = cvt8(itrow + khi - 128); }
        else                { c1v = *(const v8bf*)(alt + khi - 256); }
        zf[ks] = packA(c0, c1v);
    }
    int nt = wid;
    int c = nt * 16 + m;
    { // P2 = z_prefix @ W2[:,0:384].T + b2
        v8f acc = {};
#pragma unroll
        for (int ks = 0; ks < 12; ks++) acc = wmma_bf(zf[ks], bfrag(W2b, 512, 0, nt, ks, lane), acc);
        float bc = b2[c];
#pragma unroll
        for (int j = 0; j < 8; j++) {
            int rr = rt * 16 + j + half8; int bb = rr / TT, tt = rr % TT;
            P2[(bb * TT + tt) * DK + c] = acc[j] + bc;
        }
    }
    { // P3 = z_prefix @ W3[:,0:384].T + b3
        v8f acc = {};
#pragma unroll
        for (int ks = 0; ks < 12; ks++) acc = wmma_bf(zf[ks], bfrag(W3b, 512, 0, nt, ks, lane), acc);
        float bc = b3[c];
#pragma unroll
        for (int j = 0; j < 8; j++) {
            int rr = rt * 16 + j + half8; int bb = rr / TT, tt = rr % TT;
            P3[(bb * TT + tt) * DK + c] = acc[j] + bc;
        }
    }
    { // P4 = it_emb @ W4[:,256:384].T + b4
        v8f acc = {};
#pragma unroll
        for (int ks = 0; ks < 4; ks++) {
            int klo = ks * 32 + half8;
            v16bf a = packA(cvt8(itrow + klo), cvt8(itrow + klo + 16));
            acc = wmma_bf(a, bfrag(W4b, 384, 256, nt, ks, lane), acc);
        }
        float bc = b4[c];
#pragma unroll
        for (int j = 0; j < 8; j++) {
            int rr = rt * 16 + j + half8; int bb = rr / TT, tt = rr % TT;
            P4[(bb * TT + tt) * DK + c] = acc[j] + bc;
        }
    }
    { // E5 = e_emb[t+1] @ W5[:,0:128].T + b5
        v8f acc = {};
#pragma unroll
        for (int ks = 0; ks < 4; ks++) {
            int klo = ks * 32 + half8;
            v16bf a = packA(cvt8(enrow + klo), cvt8(enrow + klo + 16));
            acc = wmma_bf(a, bfrag(W5b, 256, 0, nt, ks, lane), acc);
        }
        float bc = b5[c];
#pragma unroll
        for (int j = 0; j < 8; j++) {
            int rr = rt * 16 + j + half8; int bb = rr / TT, tt = rr % TT;
            E5[(bb * TT + tt) * DK + c] = acc[j] + bc;
        }
    }
}

// ---------------- K3: persistent sequential scan (8 WGs x 16 batch rows) ----------------
__global__ __launch_bounds__(256) void k3_scan(
    const int* __restrict__ e_data, const float* __restrict__ q_matrix,
    const __bf16* __restrict__ W2b, const __bf16* __restrict__ W3b,
    const __bf16* __restrict__ W4b, const __bf16* __restrict__ W5b,
    const float* __restrict__ P2, const float* __restrict__ P3,
    const float* __restrict__ P4, const float* __restrict__ E5,
    float* __restrict__ h, float* __restrict__ pred)
{
    extern __shared__ char smem[];
    __bf16* sW2d = (__bf16*)smem;          // 128x128 each, bf16
    __bf16* sW3d = sW2d + 16384;
    __bf16* sW4a = sW3d + 16384;
    __bf16* sW4b = sW4a + 16384;
    __bf16* sW5h = sW4b + 16384;
    float*  sLG   = (float*)(sW5h + 16384); // 16x128 f32 buffers
    float*  sLGW4 = sLG + 2048;
    float*  sHT   = sLGW4 + 2048;
    float*  sU2   = sHT + 2048;
    float*  sU3   = sU2 + 2048;
    float*  sU5   = sU3 + 2048;
    float*  sQe   = sU5 + 2048;            // 16x101
    float*  sQn   = sQe + 16 * NQ;         // 16x101
    float*  sQs   = sQn + 16 * NQ;         // 16
    __bf16* sLGb  = (__bf16*)(sQs + 16);   // 16x128 bf16
    __bf16* sHTb  = sLGb + 2048;           // 16x128 bf16

    int tid = threadIdx.x;
    int lane = tid & 31, wid = tid >> 5;
    int half8 = (lane & 16) ? 8 : 0;
    int m = lane & 15;
    int bbase = blockIdx.x * 16;

    // Stage scan weights into LDS once (reused for all 127 steps)
    for (int i = tid; i < 2048; i += 256) {
        int n = i >> 4, kc = (i & 15) << 3;
        *(uint4*)(sW2d + n * 128 + kc) = *(const uint4*)(W2b + n * 512 + 384 + kc);
        *(uint4*)(sW3d + n * 128 + kc) = *(const uint4*)(W3b + n * 512 + 384 + kc);
        *(uint4*)(sW4a + n * 128 + kc) = *(const uint4*)(W4b + n * 384 + kc);
        *(uint4*)(sW4b + n * 128 + kc) = *(const uint4*)(W4b + n * 384 + 128 + kc);
        *(uint4*)(sW5h + n * 128 + kc) = *(const uint4*)(W5b + n * 256 + 128 + kc);
    }
    // Initial h_tilde from q0 and h(=h0 broadcast)
    if (tid < 16) sQs[tid] = 0.f;
    __syncthreads();
    for (int i = tid; i < 16 * NQ; i += 256) {
        int b = i / NQ, n = i % NQ;
        float v = q_matrix[e_data[(bbase + b) * SS + 0] * NQ + n];
        sQe[i] = v;
        atomicAdd(&sQs[b], v);
    }
    __syncthreads();
    for (int i = tid; i < 2048; i += 256) {
        int b = i >> 7, cx = i & 127;
        const float* hb = h + ((size_t)(bbase + b) * NQ) * DK + cx;
        float s = 0.f;
        for (int n = 0; n < NQ; n++) s += sQe[b * NQ + n] * hb[n * DK];
        sHTb[i] = (__bf16)(s / sQs[b]);
    }

    for (int t = 0; t < TT; t++) {
        __syncthreads();
        for (int i = tid; i < 2048; i += 256) sHT[i] = 0.f;
        if (tid < 16) sQs[tid] = 0.f;
        __syncthreads();
        for (int i = tid; i < 16 * NQ; i += 256) {
            int b = i / NQ, n = i % NQ;
            sQe[i] = q_matrix[e_data[(bbase + b) * SS + t] * NQ + n];
            float v = q_matrix[e_data[(bbase + b) * SS + t + 1] * NQ + n];
            sQn[i] = v;
            atomicAdd(&sQs[b], v);
        }
        __syncthreads();
        // Phase A: u2 = P2 + h_tilde@W2d.T ; u3 = P3 + h_tilde@W3d.T   (16 jobs on 8 waves)
        {
            v16bf af[4];
#pragma unroll
            for (int ks = 0; ks < 4; ks++) af[ks] = afrag_bf(sHTb, 128, ks, lane);
#pragma unroll
            for (int jj = 0; jj < 2; jj++) {
                int job = wid + jj * 8;
                int g = job >> 3, nt = job & 7;
                const __bf16* W = g ? sW3d : sW2d;
                const float*  P = g ? P3 : P2;
                float*        U = g ? sU3 : sU2;
                v8f acc = {};
#pragma unroll
                for (int ks = 0; ks < 4; ks++) acc = wmma_bf(af[ks], bfrag(W, 128, 0, nt, ks, lane), acc);
                int c = nt * 16 + m;
#pragma unroll
                for (int j = 0; j < 8; j++) {
                    int bl = j + half8;
                    U[bl * 128 + c] = acc[j] + P[((bbase + bl) * TT + t) * DK + c];
                }
            }
        }
        __syncthreads();
        for (int i = tid; i < 2048; i += 256) {
            float lg = sigf(sU3[i]) * (tanhf(sU2[i]) + 1.0f) * 0.5f;
            sLG[i] = lg;
            sLGb[i] = (__bf16)lg;
        }
        __syncthreads();
        // Phase B: LGW4 = LG@W4b.T + P4
        {
            v16bf af[4];
#pragma unroll
            for (int ks = 0; ks < 4; ks++) af[ks] = afrag_bf(sLGb, 128, ks, lane);
            int nt = wid;
            v8f acc = {};
#pragma unroll
            for (int ks = 0; ks < 4; ks++) acc = wmma_bf(af[ks], bfrag(sW4b, 128, 0, nt, ks, lane), acc);
            int c = nt * 16 + m;
#pragma unroll
            for (int j = 0; j < 8; j++) {
                int bl = j + half8;
                sLGW4[bl * 128 + c] = acc[j] + P4[((bbase + bl) * TT + t) * DK + c];
            }
        }
        __syncthreads();
        // Phase C: per-n: gamma_f = sigmoid(h_pre@W4a.T + LGW4); h = q_e*LG + gamma_f*h_pre;
        //          accumulate q_next*h into register partials for h_tilde.
        {
            float vacc[8][8];
#pragma unroll
            for (int a = 0; a < 8; a++)
#pragma unroll
                for (int bq = 0; bq < 8; bq++) vacc[a][bq] = 0.f;
            for (int n = wid; n < NQ; n += 8) {
                const float* hrow = h + ((size_t)(bbase + m) * NQ + n) * DK;
                v16bf af[4];
#pragma unroll
                for (int ks = 0; ks < 4; ks++) {
                    int klo = ks * 32 + half8;
                    af[ks] = packA(cvt8(hrow + klo), cvt8(hrow + klo + 16));
                }
                float qe_j[8], qn_j[8];
#pragma unroll
                for (int j = 0; j < 8; j++) {
                    int bl = j + half8;
                    qe_j[j] = sQe[bl * NQ + n];
                    qn_j[j] = sQn[bl * NQ + n];
                }
#pragma unroll
                for (int nt = 0; nt < 8; nt++) {
                    v8f acc = {};
#pragma unroll
                    for (int ks = 0; ks < 4; ks++) acc = wmma_bf(af[ks], bfrag(sW4a, 128, 0, nt, ks, lane), acc);
                    int c = nt * 16 + m;
#pragma unroll
                    for (int j = 0; j < 8; j++) {
                        int bl = j + half8;
                        float* hp_ptr = h + (((size_t)(bbase + bl) * NQ + n) * DK + c);
                        float hp = *hp_ptr;
                        float gf = sigf(acc[j] + sLGW4[bl * 128 + c]);
                        float hn = qe_j[j] * sLG[bl * 128 + c] + gf * hp;
                        *hp_ptr = hn;
                        vacc[nt][j] += qn_j[j] * hn;
                    }
                }
            }
#pragma unroll
            for (int nt = 0; nt < 8; nt++) {
                int c = nt * 16 + m;
#pragma unroll
                for (int j = 0; j < 8; j++)
                    atomicAdd(&sHT[(j + half8) * 128 + c], vacc[nt][j]);
            }
        }
        __syncthreads();
        for (int i = tid; i < 2048; i += 256) {
            int b = i >> 7;
            sHTb[i] = (__bf16)(sHT[i] / sQs[b]);
        }
        __syncthreads();
        // Phase D: y = mean(sigmoid(h_tilde@W5h.T + E5))
        {
            v16bf af[4];
#pragma unroll
            for (int ks = 0; ks < 4; ks++) af[ks] = afrag_bf(sHTb, 128, ks, lane);
            int nt = wid;
            v8f acc = {};
#pragma unroll
            for (int ks = 0; ks < 4; ks++) acc = wmma_bf(af[ks], bfrag(sW5h, 128, 0, nt, ks, lane), acc);
            int c = nt * 16 + m;
#pragma unroll
            for (int j = 0; j < 8; j++) {
                int bl = j + half8;
                sU5[bl * 128 + c] = sigf(acc[j] + E5[((bbase + bl) * TT + t) * DK + c]);
            }
        }
        __syncthreads();
        if (tid < 16) {
            float s = 0.f;
            for (int c2 = 0; c2 < 128; c2++) s += sU5[tid * 128 + c2];
            pred[(bbase + tid) * SS + t + 1] = s * (1.0f / 128.0f);
            if (t == 0) pred[(bbase + tid) * SS] = 0.0f;
        }
    }
}

// ---------------- host launcher ----------------
extern "C" void kernel_launch(void* const* d_in, const int* in_sizes, int n_in,
                              void* d_out, int out_size, void* d_ws, size_t ws_size,
                              hipStream_t stream)
{
    const int*   e_data   = (const int*)d_in[0];
    const int*   a_data   = (const int*)d_in[1];
    const int*   it_data  = (const int*)d_in[2];
    const int*   at_data  = (const int*)d_in[3];
    const float* q_matrix = (const float*)d_in[4];
    const float* e_w      = (const float*)d_in[5];
    const float* at_w     = (const float*)d_in[6];
    const float* it_w     = (const float*)d_in[7];
    const float* h0       = (const float*)d_in[8];
    const float* W1 = (const float*)d_in[9];  const float* b1 = (const float*)d_in[10];
    const float* W2 = (const float*)d_in[11]; const float* b2 = (const float*)d_in[12];
    const float* W3 = (const float*)d_in[13]; const float* b3 = (const float*)d_in[14];
    const float* W4 = (const float*)d_in[15]; const float* b4 = (const float*)d_in[16];
    const float* W5 = (const float*)d_in[17]; const float* b5 = (const float*)d_in[18];
    float* pred = (float*)d_out;
    char* ws = (char*)d_ws;

    size_t off = 0;
    auto carve = [&](size_t bytes) -> char* {
        char* p = ws + off;
        off += (bytes + 255) & ~(size_t)255;
        return p;
    };
    __bf16* AL  = (__bf16*)carve((size_t)BB * SS * DK * 2);
    __bf16* W1b = (__bf16*)carve(128 * 384 * 2);
    __bf16* W2b = (__bf16*)carve(128 * 512 * 2);
    __bf16* W3b = (__bf16*)carve(128 * 512 * 2);
    __bf16* W4b = (__bf16*)carve(128 * 384 * 2);
    __bf16* W5b = (__bf16*)carve(128 * 256 * 2);
    float*  c1  = (float*)carve(128 * 4);
    float*  P2  = (float*)carve((size_t)BB * TT * DK * 4);
    float*  P3  = (float*)carve((size_t)BB * TT * DK * 4);
    float*  P4  = (float*)carve((size_t)BB * TT * DK * 4);
    float*  E5  = (float*)carve((size_t)BB * TT * DK * 4);
    float*  h   = (float*)carve((size_t)BB * NQ * DK * 4);

    k_cvt<<<(128 * 384 + 255) / 256, 256, 0, stream>>>(W1, W1b, 128 * 384);
    k_cvt<<<(128 * 512 + 255) / 256, 256, 0, stream>>>(W2, W2b, 128 * 512);
    k_cvt<<<(128 * 512 + 255) / 256, 256, 0, stream>>>(W3, W3b, 128 * 512);
    k_cvt<<<(128 * 384 + 255) / 256, 256, 0, stream>>>(W4, W4b, 128 * 384);
    k_cvt<<<(128 * 256 + 255) / 256, 256, 0, stream>>>(W5, W5b, 128 * 256);
    k_c1<<<1, 128, 0, stream>>>(W1, c1);
    k_hinit<<<(BB * NQ * DK + 255) / 256, 256, 0, stream>>>(h0, h);

    k1_AL<<<BB * SS / 16, 256, 0, stream>>>(e_data, at_data, a_data, e_w, at_w, W1b, b1, c1, AL);
    k2_pre<<<BB * TT / 16, 256, 0, stream>>>(e_data, it_data, e_w, it_w, AL,
                                             W2b, W3b, W4b, W5b, b2, b3, b4, b5,
                                             P2, P3, P4, E5);

    size_t lds = (size_t)(5 * 16384 * 2)                 // weights (bf16)
               + (size_t)(6 * 2048 + 2 * 16 * NQ + 16) * 4 // f32 buffers + q rows + sums
               + (size_t)(2 * 2048 * 2);                 // bf16 LG / h_tilde
    hipFuncSetAttribute(reinterpret_cast<const void*>(k3_scan),
                        hipFuncAttributeMaxDynamicSharedMemorySize, (int)lds);
    k3_scan<<<BB / 16, 256, lds, stream>>>(e_data, q_matrix, W2b, W3b, W4b, W5b,
                                           P2, P3, P4, E5, h, pred);

    (void)in_sizes; (void)n_in; (void)out_size; (void)ws_size;
}
